// PoissonDEQ_1477468750556
// MI455X (gfx1250) — matrix operand order repaired and checked by reading, not compiled, not run on hardware
//
#include <hip/hip_runtime.h>
#include <hip/hip_bf16.h>

// ---------------------------------------------------------------------------
// PoissonDEQ forward for gfx1250 (MI455X).
//   Dense h@W maps  -> v_wmma_f32_16x16x32_bf16, 16x64 strip per wave,
//                      fully unrolled K (Kpad=128) => 16 WMMAs per wave;
//                      per K-step all 10 b128 operand loads issue before the
//                      4-WMMA chain (distinct B fragment registers) so the
//                      scheduler can overlap loads with matrix ops.
//   segment_sum     -> one wave/edge, float4 gather + global_atomic_add_f32
//                      (hot set L2-resident: 25.6MB features, 3.2MB norm)
//   log_softmax     -> wave32 shfl_xor reductions
// ---------------------------------------------------------------------------

typedef __attribute__((ext_vector_type(16))) __bf16 v16bf;
typedef __attribute__((ext_vector_type(8)))  float  v8f;

#define CPAD 128   // padded feature dim (all hop buffers are N x 128)
#define KPAD 128   // padded K for every GEMM (compile-time, K-loop unrolled)

__device__ __forceinline__ unsigned f2bf1(float f) {
  unsigned u = __float_as_uint(f);
  u += 0x7FFFu + ((u >> 16) & 1u);       // round-to-nearest-even
  return u >> 16;
}
__device__ __forceinline__ unsigned packbf(float lo, float hi) {
  return f2bf1(lo) | (f2bf1(hi) << 16);
}
__device__ __forceinline__ float gelu_exact(float x) {
  return 0.5f * x * (1.0f + erff(x * 0.70710678118654752f));
}

// ---------------------------------------------------------------- utilities
__global__ void k_zero4(float4* p, long n4) {
  long i = (long)blockIdx.x * blockDim.x + threadIdx.x;
  if (i < n4) p[i] = make_float4(0.f, 0.f, 0.f, 0.f);
}

__global__ void k_deg(const float* __restrict__ ea, const int* __restrict__ col,
                      float* __restrict__ deg, int E) {
  int e = blockIdx.x * blockDim.x + threadIdx.x;
  if (e < E) atomicAdd(&deg[col[e]], ea[e]);
}

__global__ void k_dinv(float* deg, int n) {
  int i = blockIdx.x * blockDim.x + threadIdx.x;
  if (i < n) { float d = deg[i]; deg[i] = (d > 0.f) ? rsqrtf(d) : 0.f; }
}

__global__ void k_norm(const int* __restrict__ row, const int* __restrict__ col,
                       const float* __restrict__ ea, const float* __restrict__ dinv,
                       float* __restrict__ nrm, int E) {
  int e = blockIdx.x * blockDim.x + threadIdx.x;
  if (e < E) nrm[e] = dinv[row[e]] * ea[e] * dinv[col[e]];
}

// pack stacked weights [S][Kin][Cout] f32 -> B-transposed bf16 pairs:
// dst[((s*Cout + n) * KPAD/2) + k2] = pack(W[s][2k2][n], W[s][2k2+1][n])
// => the 8-dword B fragment of a lane is two non-adjacent b128 loads.
__global__ void k_packw(const float* __restrict__ src, unsigned* __restrict__ dst,
                        int S, int Kin, int Cout) {
  long idx = (long)blockIdx.x * blockDim.x + threadIdx.x;
  long total = (long)S * Cout * (KPAD >> 1);
  if (idx >= total) return;
  int k2 = (int)(idx % (KPAD >> 1));
  int n  = (int)((idx / (KPAD >> 1)) % Cout);
  int s  = (int)(idx / ((long)Cout * (KPAD >> 1)));
  int k  = 2 * k2;
  float lo = (k     < Kin) ? src[((long)s * Kin + k    ) * Cout + n] : 0.f;
  float hi = (k + 1 < Kin) ? src[((long)s * Kin + k + 1) * Cout + n] : 0.f;
  dst[idx] = packbf(lo, hi);
}

// f32 -> packed bf16 pairs (row-major activations; adjacent K in one dword)
__global__ void k_cvt(const float* __restrict__ src, unsigned* __restrict__ dst, long npairs) {
  long p = (long)blockIdx.x * blockDim.x + threadIdx.x;
  if (p < npairs) dst[p] = packbf(src[2 * p], src[2 * p + 1]);
}

// ---------------------------------------------------------------- WMMA GEMM
// D[M x Cout] (+)= A[M x 128]_bf16 * B[128 x Cout]_bf16
// One wave per 16(M) x 64(N) strip: 4 accumulators share each A fragment.
// K fully unrolled (4 steps of 32) => 16 v_wmma per wave, b128 operand loads.
// A4:  [M][16] uint4 (row-major bf16 pairs), Bt4: [Cout][16] uint4 (K-pairs).
union Frag { v16bf v; uint4 q[2]; };

__global__ void __launch_bounds__(256) k_gemm(
    const uint4* __restrict__ A4, const uint4* __restrict__ Bt4,
    float* __restrict__ D, int M, int Cout, int accumulate)
{
  const int lane = threadIdx.x & 31;
  const int wave = blockIdx.x * (blockDim.x >> 5) + (threadIdx.x >> 5);
  const int nstrips = Cout >> 6;                  // 64-wide output strips
  const int mt = wave / nstrips;
  const int st = wave - mt * nstrips;
  if (mt * 16 >= M) return;                       // wave-uniform exit

  const int half = lane >> 4;                     // 0: lanes 0-15, 1: 16-31
  const int l15  = lane & 15;

  const int rowD0 = mt * 16 + half * 8;           // C/D layout: vgpr r -> M=r(+8)
  const int colN0 = st * 64 + l15;

  v8f acc[4];
  if (accumulate) {
#pragma unroll
    for (int nt = 0; nt < 4; ++nt)
#pragma unroll
      for (int r = 0; r < 8; ++r)
        acc[nt][r] = D[(size_t)(rowD0 + r) * Cout + colN0 + nt * 16];
  } else {
#pragma unroll
    for (int nt = 0; nt < 4; ++nt)
#pragma unroll
      for (int r = 0; r < 8; ++r) acc[nt][r] = 0.f;
  }

  const uint4* Arow = A4 + (size_t)(mt * 16 + l15) * (KPAD / 8);  // 16 uint4/row
  const uint4* Bc0  = Bt4 + (size_t)(colN0     ) * (KPAD / 8);
  const uint4* Bc1  = Bt4 + (size_t)(colN0 + 16) * (KPAD / 8);
  const uint4* Bc2  = Bt4 + (size_t)(colN0 + 32) * (KPAD / 8);
  const uint4* Bc3  = Bt4 + (size_t)(colN0 + 48) * (KPAD / 8);

#pragma unroll
  for (int s = 0; s < KPAD / 32; ++s) {
    // Issue all 10 b128 loads of this K-step before any WMMA: distinct
    // fragment registers allow partial loadcnt waits / load-WMMA overlap.
    Frag a, b0, b1, b2, b3;
    a.q[0]  = Arow[4 * s + half];
    a.q[1]  = Arow[4 * s + half + 2];
    b0.q[0] = Bc0[4 * s + half];  b0.q[1] = Bc0[4 * s + half + 2];
    b1.q[0] = Bc1[4 * s + half];  b1.q[1] = Bc1[4 * s + half + 2];
    b2.q[0] = Bc2[4 * s + half];  b2.q[1] = Bc2[4 * s + half + 2];
    b3.q[0] = Bc3[4 * s + half];  b3.q[1] = Bc3[4 * s + half + 2];
    acc[0] = __builtin_amdgcn_wmma_f32_16x16x32_bf16(
        false, a.v, false, b0.v, (short)0, acc[0], false, false);
    acc[1] = __builtin_amdgcn_wmma_f32_16x16x32_bf16(
        false, a.v, false, b1.v, (short)0, acc[1], false, false);
    acc[2] = __builtin_amdgcn_wmma_f32_16x16x32_bf16(
        false, a.v, false, b2.v, (short)0, acc[2], false, false);
    acc[3] = __builtin_amdgcn_wmma_f32_16x16x32_bf16(
        false, a.v, false, b3.v, (short)0, acc[3], false, false);
  }

#pragma unroll
  for (int nt = 0; nt < 4; ++nt)
#pragma unroll
    for (int r = 0; r < 8; ++r)
      D[(size_t)(rowD0 + r) * Cout + colN0 + nt * 16] = acc[nt][r];
}

// ---------------------------------------------------------------- SpMM scatter
// out[col[e]] += norm[e] * g[row[e]] : one wave/edge, float4 gather,
// atomics skipped on padded channels (c >= C).
__global__ void k_spmm(const float* __restrict__ g, const float* __restrict__ nrm,
                       const int* __restrict__ row, const int* __restrict__ col,
                       float* __restrict__ out, int E, int C) {
  int e = (blockIdx.x * blockDim.x + threadIdx.x) >> 5;
  int lane = threadIdx.x & 31;
  if (e >= E) return;
  const float w = nrm[e];
  const float4 v4 = ((const float4*)(g + (size_t)row[e] * CPAD))[lane];
  float* od = out + (size_t)col[e] * CPAD + 4 * lane;
  const int c0 = 4 * lane;
  if (c0     < C) atomicAdd(od + 0, w * v4.x);
  if (c0 + 1 < C) atomicAdd(od + 1, w * v4.y);
  if (c0 + 2 < C) atomicAdd(od + 2, w * v4.z);
  if (c0 + 3 < C) atomicAdd(od + 3, w * v4.w);
}

// ---------------------------------------------------------------- epilogues
// bias + relu + bf16 staging in one pass (thread per bf16 pair)
__global__ void k_bias_relu_cvt(const float* __restrict__ acc, const float* __restrict__ b,
                                float* __restrict__ g, unsigned* __restrict__ gbf,
                                int N, int Cout) {
  long idx = (long)blockIdx.x * blockDim.x + threadIdx.x;
  if (idx >= (long)N * (CPAD / 2)) return;
  int i = (int)(idx >> 6), c0 = (int)((idx & 63) << 1);
  float v0 = (c0     < Cout) ? fmaxf(acc[(size_t)i * Cout + c0    ] + b[c0    ], 0.f) : 0.f;
  float v1 = (c0 + 1 < Cout) ? fmaxf(acc[(size_t)i * Cout + c0 + 1] + b[c0 + 1], 0.f) : 0.f;
  g[(size_t)i * CPAD + c0]     = v0;
  g[(size_t)i * CPAD + c0 + 1] = v1;
  gbf[idx] = packbf(v0, v1);
}

// log_softmax over 64 channels (one wave per node, 2 values per lane)
__global__ void k_logsoftmax64(const float* __restrict__ acc, const float* __restrict__ b,
                               float* __restrict__ z, int N) {
  int i = (blockIdx.x * blockDim.x + threadIdx.x) >> 5;
  int lane = threadIdx.x & 31;
  if (i >= N) return;
  float v0 = acc[(size_t)i * 64 + lane]      + b[lane];
  float v1 = acc[(size_t)i * 64 + lane + 32] + b[lane + 32];
  float m = fmaxf(v0, v1);
#pragma unroll
  for (int off = 16; off >= 1; off >>= 1) m = fmaxf(m, __shfl_xor(m, off, 32));
  float s = expf(v0 - m) + expf(v1 - m);
#pragma unroll
  for (int off = 16; off >= 1; off >>= 1) s += __shfl_xor(s, off, 32);
  float l = m + logf(s);
  z[(size_t)i * 64 + lane]      = v0 - l;
  z[(size_t)i * 64 + lane + 32] = v1 - l;
}

// h = [z | pos | 0-pad] + bf16 staging in one pass (thread per pair)
__global__ void k_build_h_cvt(const float* __restrict__ z, const float* __restrict__ pos,
                              float* __restrict__ h, unsigned* __restrict__ gbf, int N) {
  long idx = (long)blockIdx.x * blockDim.x + threadIdx.x;
  if (idx >= (long)N * (CPAD / 2)) return;
  int i = (int)(idx >> 6), c0 = (int)((idx & 63) << 1);
  float v[2];
#pragma unroll
  for (int j = 0; j < 2; ++j) {
    int c = c0 + j;
    if (c < 64)      v[j] = z[(size_t)i * 64 + c];
    else if (c < 67) v[j] = pos[(size_t)i * 3 + (c - 64)];
    else             v[j] = 0.f;
  }
  h[(size_t)i * CPAD + c0]     = v[0];
  h[(size_t)i * CPAD + c0 + 1] = v[1];
  gbf[idx] = packbf(v[0], v[1]);
}

// ---------------------------------------------------------------- tiny MLPs
__global__ void k_up1(const float* __restrict__ x, const float* __restrict__ w1,
                      const float* __restrict__ b1, float* __restrict__ t, int N) {
  long idx = (long)blockIdx.x * blockDim.x + threadIdx.x;
  if (idx >= (long)N * 64) return;
  int i = (int)(idx >> 6), j = (int)(idx & 63);
  float s = b1[j];
#pragma unroll
  for (int k = 0; k < 4; ++k) s += x[(size_t)i * 4 + k] * w1[k * 64 + j];
  t[idx] = gelu_exact(s);
}

__global__ void k_up2(const float* __restrict__ t, const float* __restrict__ w2,
                      const float* __restrict__ b2, float* __restrict__ z, int N) {
  long idx = (long)blockIdx.x * blockDim.x + threadIdx.x;
  if (idx >= (long)N * 64) return;
  int i = (int)(idx >> 6), j = (int)(idx & 63);
  float s = b2[j];
  for (int k = 0; k < 64; ++k) s += t[(size_t)i * 64 + k] * w2[k * 64 + j];
  z[idx] = s;
}

__global__ void k_down(const float* __restrict__ z, const float* __restrict__ dw1,
                       const float* __restrict__ db1, const float* __restrict__ dw2,
                       const float* __restrict__ db2, float* __restrict__ out, int N) {
  int i = blockIdx.x * blockDim.x + threadIdx.x;
  if (i >= N) return;
  float s = db1[0];
  for (int k = 0; k < 64; ++k) s += z[(size_t)i * 64 + k] * dw1[k];
  out[i] = gelu_exact(s) * dw2[0] + db2[0];
}

// ---------------------------------------------------------------------------
extern "C" void kernel_launch(void* const* d_in, const int* in_sizes, int n_in,
                              void* d_out, int out_size, void* d_ws, size_t ws_size,
                              hipStream_t stream) {
  const float* x       = (const float*)d_in[0];
  const float* pos     = (const float*)d_in[1];
  const float* eattr   = (const float*)d_in[2];
  const float* up_w1   = (const float*)d_in[3];
  const float* up_b1   = (const float*)d_in[4];
  const float* up_w2   = (const float*)d_in[5];
  const float* up_b2   = (const float*)d_in[6];
  const float* c1_w    = (const float*)d_in[7];
  const float* c1_b    = (const float*)d_in[8];
  const float* c2_w    = (const float*)d_in[9];
  const float* c2_b    = (const float*)d_in[10];
  const float* c3_w    = (const float*)d_in[11];
  const float* c3_b    = (const float*)d_in[12];
  const float* down_w1 = (const float*)d_in[13];
  const float* down_b1 = (const float*)d_in[14];
  const float* down_w2 = (const float*)d_in[15];
  const float* down_b2 = (const float*)d_in[16];
  const int*   eidx    = (const int*)d_in[17];

  const int N = in_sizes[0] / 4;       // 50000 (multiple of 16)
  const int E = in_sizes[2];           // 800000
  const int* row = eidx;
  const int* col = eidx + E;

  // ---- workspace carve-up (~106 MB; hot set is L2-resident) ----
  size_t off = 0;
  auto take = [&](size_t bytes) -> void* {
    void* p = (char*)d_ws + off;
    off += (bytes + 255) & ~(size_t)255;
    return p;
  };
  float*    deg  = (float*)take((size_t)N * 4);
  float*    nrm  = (float*)take((size_t)E * 4);
  float*    gA   = (float*)take((size_t)N * CPAD * 4);   // hop buffer (ping)
  float*    gB   = (float*)take((size_t)N * CPAD * 4);   // hop buffer (pong)
  float*    accb = (float*)take((size_t)N * CPAD * 4);   // tag accumulator / mlp temp
  float*    zb   = (float*)take((size_t)N * 64 * 4);     // z iterate
  unsigned* gbf  = (unsigned*)take((size_t)N * 64 * 4);  // bf16-staged activations
  unsigned* w1p  = (unsigned*)take((size_t)4 * 128 * 64 * 4);  // [S][Cout][K/2]
  unsigned* w2p  = (unsigned*)take((size_t)4 * 128 * 64 * 4);
  unsigned* w3p  = (unsigned*)take((size_t)4 *  64 * 64 * 4);
  (void)ws_size; (void)n_in; (void)out_size;

  const int T = 256;
  auto nb = [](long n) { return (unsigned)((n + 255) / 256); };

  // ---- gcn_norm ----
  k_zero4<<<nb(N / 4 + 1), T, 0, stream>>>((float4*)deg, (N + 3) / 4);
  k_deg <<<nb(E), T, 0, stream>>>(eattr, col, deg, E);
  k_dinv<<<nb(N), T, 0, stream>>>(deg, N);
  k_norm<<<nb(E), T, 0, stream>>>(row, col, eattr, deg, nrm, E);

  // ---- weight packing (f32 -> bf16, B-transposed K-pair layout, K->128) ----
  k_packw<<<nb((long)4 * 128 * 64), T, 0, stream>>>(c1_w, w1p, 4,  67, 128);
  k_packw<<<nb((long)4 * 128 * 64), T, 0, stream>>>(c2_w, w2p, 4, 128, 128);
  k_packw<<<nb((long)4 *  64 * 64), T, 0, stream>>>(c3_w, w3p, 4, 128,  64);

  // ---- z0 = up-MLP(x) ----
  k_up1<<<nb((long)N * 64), T, 0, stream>>>(x, up_w1, up_b1, accb, N);
  k_up2<<<nb((long)N * 64), T, 0, stream>>>(accb, up_w2, up_b2, zb, N);
  k_build_h_cvt<<<nb((long)N * 64), T, 0, stream>>>(zb, pos, gA, gbf, N);

  // TAGConv: acc = sum_k (A^k h) @ W[k]; bf16 WMMA for the GEMMs
  auto run_tag = [&](const unsigned* Wp, int Cout, int Cspmm) {
    const int waves   = (N / 16) * (Cout / 64);
    const unsigned gb = (unsigned)((waves + 7) / 8);
    const size_t wstep = (size_t)Cout * 64;   // dwords per stacked weight slab
    k_gemm<<<gb, T, 0, stream>>>((const uint4*)gbf, (const uint4*)Wp, accb, N, Cout, 0);
    for (int k = 1; k <= 3; ++k) {
      k_zero4<<<nb((long)N * CPAD / 4), T, 0, stream>>>((float4*)gB, (long)N * CPAD / 4);
      k_spmm<<<nb((long)E * 32), T, 0, stream>>>(gA, nrm, row, col, gB, E, Cspmm);
      k_cvt <<<nb((long)N * 64), T, 0, stream>>>(gB, gbf, (long)N * 64);
      k_gemm<<<gb, T, 0, stream>>>((const uint4*)gbf, (const uint4*)(Wp + (size_t)k * wstep),
                                   accb, N, Cout, 1);
      float* t = gA; gA = gB; gB = t;
    }
  };

  // ---- 20 fixed-point iterations + 1 phantom-gradient step = 21 gnn apps ----
  for (int it = 0; it < 21; ++it) {
    // tag1: 67 -> 128, relu
    run_tag(w1p, 128, 67);
    k_bias_relu_cvt<<<nb((long)N * 64), T, 0, stream>>>(accb, c1_b, gA, gbf, N, 128);
    // tag2: 128 -> 128, relu
    run_tag(w2p, 128, 128);
    k_bias_relu_cvt<<<nb((long)N * 64), T, 0, stream>>>(accb, c2_b, gA, gbf, N, 128);
    // tag3: 128 -> 64, log_softmax -> z ; rebuild h for next application
    run_tag(w3p, 64, 128);
    k_logsoftmax64<<<nb((long)N * 32), T, 0, stream>>>(accb, c3_b, zb, N);
    k_build_h_cvt<<<nb((long)N * 64), T, 0, stream>>>(zb, pos, gA, gbf, N);
  }

  // ---- outputs: d_out = [ down-MLP(z_star)  (N) | z_star (N x 64) ] ----
  k_down<<<nb(N), T, 0, stream>>>(zb, down_w1, down_b1, down_w2, down_b2,
                                  (float*)d_out, N);
  hipMemcpyAsync((float*)d_out + N, zb, (size_t)N * 64 * sizeof(float),
                 hipMemcpyDeviceToDevice, stream);
}
